// QuantizedSpatialLocalityAttention_68032281969135
// MI455X (gfx1250) — matrix operand order repaired and checked by reading, not compiled, not run on hardware
//
#include <hip/hip_runtime.h>

// ---------------------------------------------------------------------------
// QuantizedSpatialLocalityAttention for MI455X (gfx1250, wave32, WMMA)
//   B=2, S=2048, D=1024, H=16, Dh=64, 4-bit fake-quant KV, spatial decay bias
// Pipeline: f32->bf16 cvt ; QKV WMMA GEMMs (quant fused into K/V epilogues,
//           V stored pre-transposed [B,H,Dh,S] for the PV WMMA B-fragments) ;
//           flash attention (wave-level, WMMA QK^T and PV) ; out-proj WMMA GEMM
// ---------------------------------------------------------------------------

typedef __attribute__((ext_vector_type(16))) __bf16 v16bf;
typedef __attribute__((ext_vector_type(8)))  __bf16 v8bf;
typedef __attribute__((ext_vector_type(8)))  float  v8f;

#define DMODEL 1024
#define SEQLEN 2048
#define NHEAD  16
#define HDIM   64
#define NBATCH 2

__device__ __forceinline__ v16bf cat8(v8bf a, v8bf b) {
  return __builtin_shufflevector(a, b, 0,1,2,3,4,5,6,7,8,9,10,11,12,13,14,15);
}

__device__ __forceinline__ v8f wmma_bf16(v16bf a, v16bf b, v8f c) {
  // D = A(16x32) * B(32x16) + C ; emits v_wmma_f32_16x16x32_bf16
  return __builtin_amdgcn_wmma_f32_16x16x32_bf16(
      /*neg_a=*/false, a, /*neg_b=*/false, b,
      /*c_mod=*/(short)0, c, /*reuse_a=*/false, /*reuse_b=*/false);
}

// A-fragment 16x32 bf16: lane<16 holds row m=lane, K 0..15;
// lane>=16 holds row m=lane-16, K 16..31. rowp = &A[row][k0].
__device__ __forceinline__ v16bf load_afrag(const __bf16* rowp, bool hi) {
  v8bf lo = *(const v8bf*)(rowp + (hi ? 16 : 0));
  v8bf up = *(const v8bf*)(rowp + (hi ? 24 : 8));
  return cat8(lo, up);
}

// B-fragment 32x16 bf16: lane<16 holds col n=lane, K 0..15; lane>=16 col n,
// K 16..31 (mirrors sparse-B table). p already offset to this lane's K range.
__device__ __forceinline__ v16bf load_bfrag(const __bf16* p) {
  return cat8(*(const v8bf*)p, *(const v8bf*)(p + 8));
}

// ---------------------------------------------------------------------------
__global__ void f2bf_kernel(const float* __restrict__ in,
                            __bf16* __restrict__ out, int n) {
  int i = blockIdx.x * blockDim.x + threadIdx.x;
  if (i < n) out[i] = (__bf16)in[i];
}

// ---------------------------------------------------------------------------
// C[M,N] = A[M,K](bf16) * W[N,K]^T(bf16) + bias, wave tile 16 x 64 (one head).
// MODE 0: Q  -> bf16 out [M,N], scaled by 1/sqrt(Dh)=0.125
// MODE 1: K  -> bf16 out [M,N], fused 4-bit fake-quant over the 64-wide head
// MODE 3: V  -> bf16 out [B,H,Dh,S] (transposed), fused 4-bit fake-quant
// MODE 2: O  -> f32 out [M,N] (final result)
template <int MODE>
__global__ __launch_bounds__(128)
void gemm_bf16_wmma(const __bf16* __restrict__ A, const __bf16* __restrict__ W,
                    const float* __restrict__ bias, void* __restrict__ outp,
                    int M) {
  const int K = DMODEL, N = DMODEL;
  const int lane = threadIdx.x & 31;
  const int wid  = threadIdx.x >> 5;
  const int gw   = blockIdx.x * 4 + wid;
  const int colTiles = N / 64;
  const int rt = gw / colTiles;
  const int ct = gw % colTiles;
  if (rt * 16 >= M) return;                    // wave-uniform
  const int row0 = rt * 16, col0 = ct * 64;
  const bool hi  = lane >= 16;
  const int  l16 = lane & 15;
  const int  koff = hi ? 16 : 0;

  v8f zero = {0.f,0.f,0.f,0.f,0.f,0.f,0.f,0.f};
  v8f acc[4] = {zero, zero, zero, zero};

  const __bf16* arow = A + (size_t)(row0 + l16) * K;
  for (int k0 = 0; k0 < K; k0 += 32) {
    __builtin_prefetch(arow + k0 + 128, 0, 1);   // global_prefetch_b8
    v16bf af = load_afrag(arow + k0, hi);
#pragma unroll
    for (int t = 0; t < 4; ++t) {
      const __bf16* wrow = W + (size_t)(col0 + 16 * t + l16) * K + k0 + koff;
      acc[t] = wmma_bf16(af, load_bfrag(wrow), acc[t]);
    }
  }

  float bs[4];
#pragma unroll
  for (int t = 0; t < 4; ++t) bs[t] = bias[col0 + 16 * t + l16];

  float v[4][8];
#pragma unroll
  for (int t = 0; t < 4; ++t)
#pragma unroll
    for (int r = 0; r < 8; ++r) v[t][r] = acc[t][r] + bs[t];

  if (MODE == 1 || MODE == 3) {
    // fake_quant: per token-row over all 64 head dims (tile == head)
#pragma unroll
    for (int r = 0; r < 8; ++r) {
      float a = fmaxf(fmaxf(fabsf(v[0][r]), fabsf(v[1][r])),
                      fmaxf(fabsf(v[2][r]), fabsf(v[3][r])));
      a = fmaxf(a, __shfl_xor(a, 1, 32));
      a = fmaxf(a, __shfl_xor(a, 2, 32));
      a = fmaxf(a, __shfl_xor(a, 4, 32));
      a = fmaxf(a, __shfl_xor(a, 8, 32));      // max over 16 lanes (half-wave)
      float sc  = fmaxf(a * (1.0f / 7.0f), 1e-8f);   // qmax = 2^(4-1)-1 = 7
      float inv = 1.0f / sc;
#pragma unroll
      for (int t = 0; t < 4; ++t) {
        float q = rintf(v[t][r] * inv);        // round-half-even == jnp.round
        q = fminf(7.f, fmaxf(-7.f, q));
        v[t][r] = q * sc;
      }
    }
  }

  if (MODE == 3) {
    // transposed store: Vt[b][h=ct][d=16t+l16][token]  (16 | SEQLEN so the
    // 16-row tile never crosses a batch boundary)
    const int bb   = row0 >> 11;               // row0 / SEQLEN
    const int tok0 = row0 & (SEQLEN - 1);
    __bf16* Vt = (__bf16*)outp;
#pragma unroll
    for (int t = 0; t < 4; ++t) {
      const size_t base =
          ((size_t)(bb * NHEAD + ct) * HDIM + 16 * t + l16) * SEQLEN + tok0;
#pragma unroll
      for (int r = 0; r < 8; ++r) {
        const int m = r + (hi ? 8 : 0);
        Vt[base + m] = (__bf16)v[t][r];
      }
    }
    return;
  }

#pragma unroll
  for (int t = 0; t < 4; ++t)
#pragma unroll
    for (int r = 0; r < 8; ++r) {
      const int m = r + (hi ? 8 : 0);
      const size_t idx = (size_t)(row0 + m) * N + col0 + 16 * t + l16;
      if (MODE == 2)      ((float*)outp)[idx]  = v[t][r];
      else if (MODE == 0) ((__bf16*)outp)[idx] = (__bf16)(v[t][r] * 0.125f);
      else                ((__bf16*)outp)[idx] = (__bf16)v[t][r];
    }
}

// ---------------------------------------------------------------------------
// Flash attention: one wave per (b, h, 16-query tile); 32-key blocks.
// Q pre-scaled by 1/sqrt(Dh). Spatial bias -dist2/(2*bw^2) added per element.
// Vt is pre-transposed [B,H,Dh,S] so PV B-fragments are two b128 loads.
__global__ __launch_bounds__(128)
void attn_flash_wmma(const __bf16* __restrict__ Qs, const __bf16* __restrict__ Kd,
                     const __bf16* __restrict__ Vt, const float* __restrict__ coords,
                     const unsigned char* __restrict__ mask,
                     const float* __restrict__ log_bw, __bf16* __restrict__ AO) {
  const int lane = threadIdx.x & 31;
  const int wid  = threadIdx.x >> 5;
  const int gw   = blockIdx.x * 4 + wid;     // 4096 waves total
  const int qt   = gw & 127;                 // 128 q tiles of 16
  const int h    = (gw >> 7) & 15;
  const int b    = gw >> 11;
  const int q0   = qt * 16;
  const bool hig = lane >= 16;
  const int  l16 = lane & 15;

  const float bw  = __expf(log_bw[h]);
  const float nib = -1.0f / (2.0f * bw * bw);

  // Q fragments for both K=32 steps of Dh=64
  const __bf16* qbase = Qs + (size_t)(b * SEQLEN + q0 + l16) * DMODEL + h * HDIM;
  v16bf qf0 = load_afrag(qbase, hig);
  v16bf qf1 = load_afrag(qbase + 32, hig);

  // query coords for this lane-group's 8 rows (m = r + 8*hig)
  float cqx[8], cqy[8];
  const int mbase = q0 + (hig ? 8 : 0);
#pragma unroll
  for (int r = 0; r < 8; ++r) {
    const float* c = coords + ((size_t)b * SEQLEN + mbase + r) * 2;
    cqx[r] = c[0]; cqy[r] = c[1];
  }

  float m_i[8], l_i[8];
#pragma unroll
  for (int r = 0; r < 8; ++r) { m_i[r] = -3.0e38f; l_i[r] = 0.f; }
  v8f zero = {0.f,0.f,0.f,0.f,0.f,0.f,0.f,0.f};
  v8f o[4] = {zero, zero, zero, zero};

  // per-wave P staging: 16 rows x 40 bf16 (80B row stride -> 16B-aligned chunks)
  __shared__ __align__(16) __bf16 plds[4][16 * 40];
  __bf16* myp = plds[wid];

  // Vt slice for this (b,h): rows = 64 head dims, cols = S tokens
  const __bf16* vslice = Vt + (size_t)(b * NHEAD + h) * HDIM * SEQLEN;

  for (int j0 = 0; j0 < SEQLEN; j0 += 32) {
    // ---- S = Q K^T (two 16x16 n-tiles, two K-steps each) ----
    v8f s0 = zero, s1 = zero;
    const __bf16* kp = Kd + (size_t)(b * SEQLEN + j0 + l16) * DMODEL + h * HDIM + (hig ? 16 : 0);
    s0 = wmma_bf16(qf0, load_bfrag(kp), s0);
    s0 = wmma_bf16(qf1, load_bfrag(kp + 32), s0);
    const __bf16* kp1 = kp + (size_t)16 * DMODEL;
    s1 = wmma_bf16(qf0, load_bfrag(kp1), s1);
    s1 = wmma_bf16(qf1, load_bfrag(kp1 + 32), s1);

    // ---- spatial bias + key mask (n = l16 in C layout) ----
    const float* ck0 = coords + ((size_t)b * SEQLEN + j0 + l16) * 2;
    const float ckx0 = ck0[0], cky0 = ck0[1];
    const float ckx1 = ck0[32], cky1 = ck0[33];
    const bool mk0 = mask[(size_t)b * SEQLEN + j0 + l16] != 0;
    const bool mk1 = mask[(size_t)b * SEQLEN + j0 + 16 + l16] != 0;
#pragma unroll
    for (int r = 0; r < 8; ++r) {
      float dx = cqx[r] - ckx0, dy = cqy[r] - cky0;
      s0[r] += nib * (dx * dx + dy * dy);
      dx = cqx[r] - ckx1; dy = cqy[r] - cky1;
      s1[r] += nib * (dx * dx + dy * dy);
      if (!mk0) s0[r] = -3.0e38f;
      if (!mk1) s1[r] = -3.0e38f;
    }

    // ---- online softmax (row stats per C-VGPR index r) ----
    float p0[8], p1[8];
#pragma unroll
    for (int r = 0; r < 8; ++r) {
      float mx = fmaxf(s0[r], s1[r]);
      mx = fmaxf(mx, __shfl_xor(mx, 1, 32));
      mx = fmaxf(mx, __shfl_xor(mx, 2, 32));
      mx = fmaxf(mx, __shfl_xor(mx, 4, 32));
      mx = fmaxf(mx, __shfl_xor(mx, 8, 32));
      const float nm = fmaxf(m_i[r], mx);
      const float al = __expf(m_i[r] - nm);
      m_i[r] = nm;
      p0[r] = __expf(s0[r] - nm);
      p1[r] = __expf(s1[r] - nm);
      float rs = p0[r] + p1[r];
      rs += __shfl_xor(rs, 1, 32);
      rs += __shfl_xor(rs, 2, 32);
      rs += __shfl_xor(rs, 4, 32);
      rs += __shfl_xor(rs, 8, 32);
      l_i[r] = l_i[r] * al + rs;
#pragma unroll
      for (int t = 0; t < 4; ++t) o[t][r] *= al;
    }

    // ---- P: C layout -> A fragment via per-wave LDS transpose ----
#pragma unroll
    for (int r = 0; r < 8; ++r) {
      const int m = r + (hig ? 8 : 0);
      myp[m * 40 + l16]      = (__bf16)p0[r];
      myp[m * 40 + 16 + l16] = (__bf16)p1[r];
    }
    const __bf16* pr = myp + l16 * 40 + (hig ? 16 : 0);
    v16bf pf = cat8(*(const v8bf*)pr, *(const v8bf*)(pr + 8));   // per-wave DS in-order

    // ---- O += P (16x32) * V (32x64): B-frags contiguous from Vt ----
#pragma unroll
    for (int t = 0; t < 4; ++t) {
      const __bf16* vp = vslice + (size_t)(16 * t + l16) * SEQLEN
                                + j0 + (hig ? 16 : 0);
      o[t] = wmma_bf16(pf, load_bfrag(vp), o[t]);
    }
  }

  // ---- normalize and store attention output (bf16 for out-proj GEMM) ----
#pragma unroll
  for (int t = 0; t < 4; ++t)
#pragma unroll
    for (int r = 0; r < 8; ++r) {
      const int m = r + (hig ? 8 : 0);
      const size_t idx = (size_t)(b * SEQLEN + q0 + m) * DMODEL + h * HDIM + 16 * t + l16;
      AO[idx] = (__bf16)(o[t][r] / l_i[r]);
    }
}

// ---------------------------------------------------------------------------
extern "C" void kernel_launch(void* const* d_in, const int* in_sizes, int n_in,
                              void* d_out, int out_size, void* d_ws, size_t ws_size,
                              hipStream_t stream) {
  const float* x      = (const float*)d_in[0];
  const float* coords = (const float*)d_in[1];
  const unsigned char* mask = (const unsigned char*)d_in[2];
  const float* Wq  = (const float*)d_in[3];
  const float* bq  = (const float*)d_in[4];
  const float* Wk  = (const float*)d_in[5];
  const float* bk  = (const float*)d_in[6];
  const float* Wv  = (const float*)d_in[7];
  const float* bv  = (const float*)d_in[8];
  const float* Wo  = (const float*)d_in[9];
  const float* bo  = (const float*)d_in[10];
  const float* lbw = (const float*)d_in[11];

  const int M = NBATCH * SEQLEN;              // 4096
  char* ws = (char*)d_ws;
  size_t off = 0;
  auto carve = [&](size_t bytes) {
    char* p = ws + off;
    off += (bytes + 255) & ~(size_t)255;
    return p;
  };
  __bf16* xb  = (__bf16*)carve((size_t)M * DMODEL * 2);
  __bf16* Wqb = (__bf16*)carve((size_t)DMODEL * DMODEL * 2);
  __bf16* Wkb = (__bf16*)carve((size_t)DMODEL * DMODEL * 2);
  __bf16* Wvb = (__bf16*)carve((size_t)DMODEL * DMODEL * 2);
  __bf16* Wob = (__bf16*)carve((size_t)DMODEL * DMODEL * 2);
  __bf16* Qs  = (__bf16*)carve((size_t)M * DMODEL * 2);
  __bf16* Kd  = (__bf16*)carve((size_t)M * DMODEL * 2);
  __bf16* Vt  = (__bf16*)carve((size_t)M * DMODEL * 2);   // [B,H,Dh,S]
  __bf16* AO  = (__bf16*)carve((size_t)M * DMODEL * 2);

  // fp32 -> bf16 conversions
  {
    int n = M * DMODEL;
    f2bf_kernel<<<n / 256, 256, 0, stream>>>(x, xb, n);
    n = DMODEL * DMODEL;
    f2bf_kernel<<<n / 256, 256, 0, stream>>>(Wq, Wqb, n);
    f2bf_kernel<<<n / 256, 256, 0, stream>>>(Wk, Wkb, n);
    f2bf_kernel<<<n / 256, 256, 0, stream>>>(Wv, Wvb, n);
    f2bf_kernel<<<n / 256, 256, 0, stream>>>(Wo, Wob, n);
  }

  const dim3 blk(128);                        // 4 waves / workgroup
  const int nblk = (M / 16) * (DMODEL / 64) / 4;   // 1024 blocks

  gemm_bf16_wmma<0><<<nblk, blk, 0, stream>>>(xb, Wqb, bq, (void*)Qs, M);
  gemm_bf16_wmma<1><<<nblk, blk, 0, stream>>>(xb, Wkb, bk, (void*)Kd, M);
  gemm_bf16_wmma<3><<<nblk, blk, 0, stream>>>(xb, Wvb, bv, (void*)Vt, M);

  attn_flash_wmma<<<nblk, blk, 0, stream>>>(Qs, Kd, Vt, coords, mask, lbw, AO);

  gemm_bf16_wmma<2><<<nblk, blk, 0, stream>>>(AO, Wob, bo, d_out, M);

  (void)in_sizes; (void)n_in; (void)out_size; (void)ws_size;
}
